// xGN_33663953666896
// MI455X (gfx1250) — matrix operand order repaired and checked by reading, not compiled, not run on hardware
//
#include <hip/hip_runtime.h>

typedef __attribute__((ext_vector_type(16))) __bf16 v16bf;
typedef __attribute__((ext_vector_type(8)))  __bf16 v8bf;
typedef __attribute__((ext_vector_type(8)))  float  v8f;

#define BS    8
#define CH    256
#define TT    2048
#define NROWS (BS * TT)
#define KNB   10

#if __has_builtin(__builtin_amdgcn_tensor_load_to_lds) && __has_builtin(__builtin_amdgcn_s_wait_tensorcnt)
#define USE_TDM 1
#else
#define USE_TDM 0
#endif

// ---------- helpers ----------

__device__ __forceinline__ unsigned short f2bf(float f) {
  unsigned int u = __float_as_uint(f);
  u += 0x7FFFu + ((u >> 16) & 1u);         // round-to-nearest-even
  return (unsigned short)(u >> 16);
}

// A fragment (16x32 bf16), row-major source with row stride `stride` (elems).
// Contiguous-k layout (same as load_b_frag): lane m / m+16 holds row m.
//   lanes 0-15 : halves 0-15 -> K=c0+0..15
//   lanes 16-31: halves 0-15 -> K=c0+16..31
__device__ __forceinline__ v16bf load_a_frag_row(const unsigned short* mat, int stride,
                                                 int rowlane, int c0, int lane) {
  int g = (lane >> 4) & 1;
  const unsigned short* p = mat + rowlane * stride + c0 + g * 16;
  v8bf lo = *reinterpret_cast<const v8bf*>(p);
  v8bf hi = *reinterpret_cast<const v8bf*>(p + 8);
  v16bf a;
#pragma unroll
  for (int i = 0; i < 8; ++i) { a[i] = lo[i]; a[i + 8] = hi[i]; }
  return a;
}

// B fragment (32x16 bf16): B[k,n] = mat[col0+n][c0+k] (mat row-major, stride elems).
// lane n / n+16 holds column n; lanes 0-15: K=c0..c0+15, lanes 16-31: K=c0+16..c0+31.
__device__ __forceinline__ v16bf load_b_frag(const unsigned short* mat, int stride,
                                             int col0, int c0, int lane) {
  int n = lane & 15; int g = (lane >> 4) & 1;
  const unsigned short* p = mat + (col0 + n) * stride + c0 + g * 16;
  return *reinterpret_cast<const v16bf*>(p);
}

__device__ __forceinline__ v8f wmma_bf16(v16bf a, v16bf b, v8f c) {
  return __builtin_amdgcn_wmma_f32_16x16x32_bf16(false, a, false, b, (short)0, c,
                                                 false, false);
}

#if USE_TDM
typedef unsigned int v4ui __attribute__((ext_vector_type(4)));
typedef int v8i_ __attribute__((ext_vector_type(8)));
typedef int v4i_ __attribute__((ext_vector_type(4)));

// TDM: DMA a 16-row x 512-byte tile (16 x 256 bf16, row stride 512B) from global
// to LDS. D# built per CDNA5 ISA ch.8: 2D tensor, 4-byte elements, 128-dword rows.
// This toolchain exposes the 6-arg builtin: (g0, g1, g2, g3, g_extra, cpol).
__device__ __forceinline__ void tdm_load_tile16(const unsigned short* gsrc, unsigned lds_off) {
  unsigned long long ga = (unsigned long long)(size_t)gsrc;
  v4ui g0;
  g0[0] = 1u;                                              // count=1, user mode
  g0[1] = lds_off;                                         // lds_addr (bytes)
  g0[2] = (unsigned)(ga & 0xffffffffu);                    // global_addr[31:0]
  g0[3] = (unsigned)((ga >> 32) & 0x1ffffffu) | 0x80000000u; // ga[56:32] | type=2
  v8i_ g1;
  g1[0] = 0x20000;                 // workgroup_mask=0, data_size=2 (4 bytes)
  g1[1] = (int)(128u << 16);       // tensor_dim0 = 128 dwords (bits 79:48 lo16)
  g1[2] = (int)(2048u << 16);      // tensor_dim0 hi16=0 | tensor_dim1 lo16 = 2048
  g1[3] = (int)(128u << 16);       // tensor_dim1 hi16=0 | tile_dim0 = 128 dwords
  g1[4] = 16;                      // tile_dim1 = 16 rows, tile_dim2 = 0
  g1[5] = 128;                     // tensor_dim0_stride = 128 dwords (lo32)
  g1[6] = (int)(128u << 16);       // stride hi16=0 | tensor_dim1_stride lo16 = 128
  g1[7] = 0;                       // tensor_dim1_stride hi bits = 0
  v4i_ z4 = {};
  v8i_ z8 = {};
  __builtin_amdgcn_tensor_load_to_lds(g0, g1, z4, z4, z8, 0);
}
#endif

// ---------- prep kernels ----------

// featb[b][t][c] = bf16(x[b][c][t])
__global__ void cvt_feat_kernel(const float* __restrict__ x, unsigned short* __restrict__ featb) {
  int id = blockIdx.x * blockDim.x + threadIdx.x;      // BS*TT*CH
  int c = id & 255; int t = (id >> 8) & 2047; int b = id >> 19;
  featb[id] = f2bf(x[(b * CH + c) * TT + t]);
}

// sq[b][t] = sum_c x^2  (f32, coalesced over t)
__global__ void sq_kernel(const float* __restrict__ x, float* __restrict__ sq) {
  int id = blockIdx.x * blockDim.x + threadIdx.x;      // BS*TT
  int t = id & 2047; int b = id >> 11;
  float acc = 0.f;
  for (int c = 0; c < CH; ++c) {
    float f = x[(b * CH + c) * TT + t];
    acc += f * f;
  }
  sq[id] = acc;
}

// split mlp_w (256 x 512) -> wnb (256x256 bf16), wcb (256x256 bf16)
__global__ void pack_mlp_kernel(const float* __restrict__ mlp_w,
                                unsigned short* __restrict__ wnb,
                                unsigned short* __restrict__ wcb) {
  int id = blockIdx.x * blockDim.x + threadIdx.x;      // 256*512
  int o = id >> 9; int c = id & 511;
  unsigned short v = f2bf(mlp_w[id]);
  if (c < 256) wnb[o * 256 + c] = v;
  else         wcb[o * 256 + (c - 256)] = v;
}

// conv_w (256,256,3) -> wtau[tau][o][i] bf16, 3 x 256x256
__global__ void pack_conv_kernel(const float* __restrict__ conv_w,
                                 unsigned short* __restrict__ wtau) {
  int id = blockIdx.x * blockDim.x + threadIdx.x;      // 3*256*256
  int tau = id >> 16; int r = id & 65535;
  int o = r >> 8; int i = r & 255;
  wtau[id] = f2bf(conv_w[(o * 256 + i) * 3 + tau]);
}

// ---------- conv1d via WMMA:  cout[b][t][o] = sum_tau sum_i feat[t-1+tau,i]*Wtau[o,i] + cb[o]
__global__ void conv_wmma_kernel(const unsigned short* __restrict__ featb,
                                 const unsigned short* __restrict__ wtau,
                                 const float* __restrict__ conv_b,
                                 float* __restrict__ cout) {
  int blk = blockIdx.x;                 // BS * 128 * 16
  int o_t = blk & 15; int t_t = (blk >> 4) & 127; int b = blk >> 11;
  int t0 = t_t * 16, o0 = o_t * 16;
  int lane = threadIdx.x & 31;
  int m = lane & 15, g = (lane >> 4) & 1;
  const unsigned short* fb = featb + (size_t)b * TT * CH;

  v16bf az;
#pragma unroll
  for (int i = 0; i < 16; ++i) az[i] = (__bf16)0.0f;

  v8f acc = {};
  for (int tau = 0; tau < 3; ++tau) {
    int trow = t0 + m - 1 + tau;
    bool ok = (trow >= 0) && (trow < TT);
    const unsigned short* wt = wtau + tau * 65536;
#pragma unroll
    for (int k = 0; k < 8; ++k) {
      v16bf a = ok ? load_a_frag_row(fb, CH, trow, k * 32, lane) : az;
      v16bf bfr = load_b_frag(wt, CH, o0, k * 32, lane);
      acc = wmma_bf16(a, bfr, acc);
    }
  }
  int n = lane & 15;
  float cb = conv_b[o0 + n];
#pragma unroll
  for (int r = 0; r < 8; ++r) {
    int mm = r + 8 * g;
    cout[((size_t)b * TT + t0 + mm) * CH + o0 + n] = acc[r] + cb;
  }
}

// ---------- H / Hc via WMMA: H[n][o] = feat @ Wn^T ; Hc = feat @ Wc^T + mlp_b
__global__ void h_wmma_kernel(const unsigned short* __restrict__ featb,
                              const unsigned short* __restrict__ wnb,
                              const unsigned short* __restrict__ wcb,
                              const float* __restrict__ mlp_b,
                              float* __restrict__ H, float* __restrict__ Hc) {
  int blk = blockIdx.x;                 // (NROWS/16) * 16
  int o_t = blk & 15; int n_t = blk >> 4;
  int n0 = n_t * 16, o0 = o_t * 16;
  int lane = threadIdx.x & 31;
  int m = lane & 15, g = (lane >> 4) & 1;

  v8f accn = {}, accc = {};
#pragma unroll
  for (int k = 0; k < 8; ++k) {
    v16bf a  = load_a_frag_row(featb, CH, n0 + m, k * 32, lane);
    v16bf bn = load_b_frag(wnb, CH, o0, k * 32, lane);
    v16bf bc = load_b_frag(wcb, CH, o0, k * 32, lane);
    accn = wmma_bf16(a, bn, accn);
    accc = wmma_bf16(a, bc, accc);
  }
  int n = lane & 15;
  float mb = mlp_b[o0 + n];
#pragma unroll
  for (int r = 0; r < 8; ++r) {
    int mm = r + 8 * g;
    H [((size_t)n0 + mm) * CH + o0 + n] = accn[r];
    Hc[((size_t)n0 + mm) * CH + o0 + n] = accc[r] + mb;
  }
}

// ---------- gram + top-k via WMMA (one wave per 16-row tile of one batch)
// dist(i,j) ranking-equivalent form: sq[j] - 2*gram[i,j]  (sq[i] constant per row)
// B tiles are staged to LDS by the Tensor Data Mover, double-buffered so the DMA
// of tile j+1 overlaps the 8 WMMAs of tile j (s_wait_tensorcnt 1 gates consumption).
__global__ void gram_topk_kernel(const unsigned short* __restrict__ featb,
                                 const float* __restrict__ sq,
                                 int* __restrict__ idxb, float* __restrict__ wb) {
  __shared__ float dist[16 * 512];
  __shared__ float lval[16][KNB];
  __shared__ int   lidx[16][KNB];
#if USE_TDM
  __shared__ __align__(32) unsigned short btile[2][16 * CH];   // 2 x 8KB double buffer
#endif

  int blk = blockIdx.x;                  // BS * 128
  int i_t = blk & 127; int b = blk >> 7;
  int i0 = i_t * 16;
  int lane = threadIdx.x & 31;
  int m = lane & 15, g = (lane >> 4) & 1;
  const unsigned short* fb = featb + (size_t)b * TT * CH;
  const float* sqb = sq + b * TT;
  const float INF = __builtin_inff();

  // preload all A fragments for this 16-row tile (8 k-chunks)
  v16bf afrag[8];
#pragma unroll
  for (int k = 0; k < 8; ++k) afrag[k] = load_a_frag_row(fb, CH, i0 + m, k * 32, lane);

  for (int q = lane; q < 16 * KNB; q += 32) { lval[q / KNB][q % KNB] = INF; lidx[q / KNB][q % KNB] = 0; }
  __syncthreads();

#if USE_TDM
  const unsigned lds0 = (unsigned)(size_t)&btile[0][0];
  const unsigned lds1 = (unsigned)(size_t)&btile[1][0];
#endif

  for (int jc = 0; jc < 4; ++jc) {       // 4 chunks of 512 columns
#if USE_TDM
    tdm_load_tile16(fb + (size_t)(jc * 512) * CH, lds0);     // prologue: stage tile 0
#endif
    for (int jt = 0; jt < 32; ++jt) {
      int j0 = jc * 512 + jt * 16;
#if USE_TDM
      if (jt + 1 < 32) {                 // prefetch next tile into other buffer
        tdm_load_tile16(fb + (size_t)(j0 + 16) * CH, ((jt + 1) & 1) ? lds1 : lds0);
        __builtin_amdgcn_s_wait_tensorcnt(1);   // current tile resident (in-order TDM)
      } else {
        __builtin_amdgcn_s_wait_tensorcnt(0);
      }
      const unsigned short* bt = (jt & 1) ? &btile[1][0] : &btile[0][0];
#endif
      v8f acc = {};
#pragma unroll
      for (int k = 0; k < 8; ++k) {
#if USE_TDM
        v16bf bfr = load_b_frag(bt, CH, 0, k * 32, lane);    // ds_load from staged tile
#else
        v16bf bfr = load_b_frag(fb, CH, j0, k * 32, lane);
#endif
        acc = wmma_bf16(afrag[k], bfr, acc);
      }
      int n = lane & 15;
      float s = sqb[j0 + n];
#pragma unroll
      for (int r = 0; r < 8; ++r) {
        int mm = r + 8 * g;
        dist[mm * 512 + jt * 16 + n] = s - 2.0f * acc[r];
      }
    }
    __syncthreads();

    // merge this chunk into each row's running top-10 (sorted ascending)
    for (int row = 0; row < 16; ++row) {
      float* drow = &dist[row * 512];
      for (int e = 0; e < KNB; ++e) {
        float mv = INF; int mi = 0x7FFFFFFF;
        for (int c = lane; c < 512; c += 32) {
          float v = drow[c];
          if (v < mv || (v == mv && c < mi)) { mv = v; mi = c; }
        }
#pragma unroll
        for (int off = 16; off > 0; off >>= 1) {
          float ov = __shfl_xor(mv, off, 32);
          int   oi = __shfl_xor(mi, off, 32);
          if (ov < mv || (ov == mv && oi < mi)) { mv = ov; mi = oi; }
        }
        if (mv >= lval[row][KNB - 1]) break;   // uniform: nothing better remains
        if (lane == 0) {
          int p = KNB - 1;
          while (p > 0 && lval[row][p - 1] > mv) {
            lval[row][p] = lval[row][p - 1];
            lidx[row][p] = lidx[row][p - 1];
            --p;
          }
          lval[row][p] = mv;
          lidx[row][p] = jc * 512 + mi;
          drow[mi] = INF;                       // remove from chunk
        }
        __syncthreads();
      }
      __syncthreads();
    }
    __syncthreads();
  }

  // emit indices + cosine weights: gram = (sq[j] - dist)/2 ; w = gram/(|fi||fj|)
  for (int q = lane; q < 16 * KNB; q += 32) {
    int row = q / KNB, k = q % KNB;
    int j = lidx[row][k];
    float dv = lval[row][k];
    float sj = sqb[j];
    float gij = 0.5f * (sj - dv);
    float si = sqb[i0 + row];
    float wv = gij * rsqrtf(si * sj);
    int nrow = b * TT + i0 + row;
    idxb[nrow * KNB + k] = j;
    wb[nrow * KNB + k] = wv;
  }
}

// ---------- final fuse: relu(conv + max_k((H[j]+Hc)*w)) + pairwise max-pool over T
__global__ void fuse_kernel(const float* __restrict__ cout, const float* __restrict__ H,
                            const float* __restrict__ Hc, const int* __restrict__ idxb,
                            const float* __restrict__ wb, float* __restrict__ out) {
  int id = blockIdx.x * blockDim.x + threadIdx.x;   // BS * 64 * 1024
  int tp = id & 1023; int o4 = (id >> 10) & 63; int b = id >> 16;
  float m0 = -__builtin_inff(), m1 = m0, m2 = m0, m3 = m0;
  for (int dt = 0; dt < 2; ++dt) {
    int t = 2 * tp + dt;
    size_t n = (size_t)b * TT + t;
    const float4 hc = *reinterpret_cast<const float4*>(Hc + n * CH + o4 * 4);
    float g0 = -__builtin_inff(), g1 = g0, g2 = g0, g3 = g0;
    for (int k = 0; k < KNB; ++k) {
      int j = idxb[n * KNB + k];
      float wv = wb[n * KNB + k];
      const float4 h = *reinterpret_cast<const float4*>(H + ((size_t)b * TT + j) * CH + o4 * 4);
      g0 = fmaxf(g0, (h.x + hc.x) * wv);
      g1 = fmaxf(g1, (h.y + hc.y) * wv);
      g2 = fmaxf(g2, (h.z + hc.z) * wv);
      g3 = fmaxf(g3, (h.w + hc.w) * wv);
    }
    const float4 cv = *reinterpret_cast<const float4*>(cout + n * CH + o4 * 4);
    m0 = fmaxf(m0, fmaxf(cv.x + g0, 0.f));
    m1 = fmaxf(m1, fmaxf(cv.y + g1, 0.f));
    m2 = fmaxf(m2, fmaxf(cv.z + g2, 0.f));
    m3 = fmaxf(m3, fmaxf(cv.w + g3, 0.f));
  }
  size_t obase = ((size_t)b * CH + o4 * 4) * (TT / 2) + tp;
  out[obase]          = m0;
  out[obase + 1024]   = m1;
  out[obase + 2048]   = m2;
  out[obase + 3072]   = m3;
}

// ---------- host ----------

extern "C" void kernel_launch(void* const* d_in, const int* in_sizes, int n_in,
                              void* d_out, int out_size, void* d_ws, size_t ws_size,
                              hipStream_t stream) {
  const float* x      = (const float*)d_in[0];
  // d_in[1] = num_frms (unused by the reference)
  const float* conv_w = (const float*)d_in[2];
  const float* conv_b = (const float*)d_in[3];
  const float* mlp_w  = (const float*)d_in[4];
  const float* mlp_b  = (const float*)d_in[5];
  float* out = (float*)d_out;

  char* ws = (char*)d_ws;
  size_t off = 0;
  auto take = [&](size_t bytes) { char* p = ws + off; off = (off + bytes + 255) & ~(size_t)255; return p; };

  unsigned short* featb = (unsigned short*)take((size_t)NROWS * CH * 2);   // 8 MB
  unsigned short* wnb   = (unsigned short*)take(256 * 256 * 2);
  unsigned short* wcb   = (unsigned short*)take(256 * 256 * 2);
  unsigned short* wtau  = (unsigned short*)take(3 * 256 * 256 * 2);
  float* sqbuf = (float*)take((size_t)NROWS * 4);
  float* cout  = (float*)take((size_t)NROWS * CH * 4);                     // 16.8 MB
  float* Hbuf  = (float*)take((size_t)NROWS * CH * 4);                     // 16.8 MB
  float* Hcbuf = (float*)take((size_t)NROWS * CH * 4);                     // 16.8 MB
  int*   idxb  = (int*)take((size_t)NROWS * KNB * 4);
  float* wbuf  = (float*)take((size_t)NROWS * KNB * 4);

  // prep
  cvt_feat_kernel<<<(NROWS * CH) / 256, 256, 0, stream>>>(x, featb);
  sq_kernel<<<NROWS / 256, 256, 0, stream>>>(x, sqbuf);
  pack_mlp_kernel<<<(256 * 512) / 256, 256, 0, stream>>>(mlp_w, wnb, wcb);
  pack_conv_kernel<<<(3 * 256 * 256) / 256, 256, 0, stream>>>(conv_w, wtau);

  // WMMA GEMMs
  conv_wmma_kernel<<<BS * 128 * 16, 32, 0, stream>>>(featb, wtau, conv_b, cout);
  h_wmma_kernel<<<(NROWS / 16) * 16, 32, 0, stream>>>(featb, wnb, wcb, mlp_b, Hbuf, Hcbuf);
  gram_topk_kernel<<<BS * 128, 32, 0, stream>>>(featb, sqbuf, idxb, wbuf);

  // fuse
  fuse_kernel<<<(BS * 64 * 1024) / 256, 256, 0, stream>>>(cout, Hbuf, Hcbuf, idxb, wbuf, out);
}
